// FlowAttention_84593675862646
// MI455X (gfx1250) — matrix operand order repaired and checked by reading, not compiled, not run on hardware
//
#include <hip/hip_runtime.h>

// ---------------- CDNA5 WMMA types / wrappers ----------------
typedef __attribute__((ext_vector_type(16))) _Float16 v16h;
typedef __attribute__((ext_vector_type(8)))  _Float16 v8h;
typedef __attribute__((ext_vector_type(8)))  float    v8f;
typedef __attribute__((ext_vector_type(4)))  int      v4i;

__device__ __forceinline__ v8f wmma_f16(v16h a, v16h b, v8f c) {
    return __builtin_amdgcn_wmma_f32_16x16x32_f16(
        false, a, false, b, (short)0, c, false, false);
}

// A-fragment (16x32, f16): lane m = lane&15; contiguous K range [(lane>>4)*16,+16)
__device__ __forceinline__ v16h load_a16x32(const _Float16* A, int lda, int lane) {
    int m   = lane & 15;
    int klo = (lane >> 4) * 16;
    v8h lo = *(const v8h*)(A + (size_t)m * lda + klo);
    v8h hi = *(const v8h*)(A + (size_t)m * lda + klo + 8);
    return __builtin_shufflevector(lo, hi, 0,1,2,3,4,5,6,7,8,9,10,11,12,13,14,15);
}

// B-fragment (32x16, f16): lane n = lane&15, contiguous K range [(lane>>4)*16,+16)
__device__ __forceinline__ v16h load_b32x16(const _Float16* Bn, int ldb, int lane) {
    int n = lane & 15;
    int k = (lane >> 4) * 16;
    return *(const v16h*)(Bn + (size_t)n * ldb + k);
}

// ---------------- gfx1250 async load-to-LDS (guarded) ----------------
#if defined(__gfx1250__) && __has_builtin(__builtin_amdgcn_global_load_async_to_lds_b128) && __has_builtin(__builtin_amdgcn_s_wait_asynccnt)
#define USE_ASYNC_LDS 1
#else
#define USE_ASYNC_LDS 0
#endif

__device__ __forceinline__ void copy16B_to_lds(const _Float16* g, _Float16* l) {
#if USE_ASYNC_LDS
    __builtin_amdgcn_global_load_async_to_lds_b128(
        (__attribute__((address_space(1))) v4i*)g,
        (__attribute__((address_space(3))) v4i*)l, 0, 0);
#else
    *(v8h*)l = *(const v8h*)g;
#endif
}
__device__ __forceinline__ void wait_async_lds() {
#if USE_ASYNC_LDS
    __builtin_amdgcn_s_wait_asynccnt(0);
#endif
}

// ---------------- problem constants ----------------
constexpr int Bsz = 4, T = 2048, D = 1024, H = 16, HD = 64;
constexpr int M_ALL = Bsz * T;            // 8192
constexpr float LOG2_THETA = 13.287712379549449f; // log2(10000)

// ---------------- 0) f32 -> f16 convert ----------------
__global__ void conv_f16_kernel(const float* __restrict__ src,
                                _Float16* __restrict__ dst, size_t n) {
    size_t i = (size_t)blockIdx.x * blockDim.x + threadIdx.x;
    size_t stride = (size_t)gridDim.x * blockDim.x;
    for (; i < n; i += stride) dst[i] = (_Float16)src[i];
}

// branch-free pipelined 16x64 GEMM body: acc0..acc3 += A(16xD) * B(4 tiles)
#define GEMM_16x64_BODY(A0, BP)                                                \
    v16h ac  = load_a16x32((A0), D, lane);                                     \
    v16h bc0 = load_b32x16((BP) + (size_t)0 * 16 * D, D, lane);                \
    v16h bc1 = load_b32x16((BP) + (size_t)1 * 16 * D, D, lane);                \
    v16h bc2 = load_b32x16((BP) + (size_t)2 * 16 * D, D, lane);                \
    v16h bc3 = load_b32x16((BP) + (size_t)3 * 16 * D, D, lane);                \
    for (int k = 0; k < D - 32; k += 32) {                                     \
        v16h an  = load_a16x32((A0) + k + 32, D, lane);                        \
        v16h bn0 = load_b32x16((BP) + (size_t)0 * 16 * D + k + 32, D, lane);   \
        v16h bn1 = load_b32x16((BP) + (size_t)1 * 16 * D + k + 32, D, lane);   \
        v16h bn2 = load_b32x16((BP) + (size_t)2 * 16 * D + k + 32, D, lane);   \
        v16h bn3 = load_b32x16((BP) + (size_t)3 * 16 * D + k + 32, D, lane);   \
        acc0 = wmma_f16(ac, bc0, acc0);                                        \
        acc1 = wmma_f16(ac, bc1, acc1);                                        \
        acc2 = wmma_f16(ac, bc2, acc2);                                        \
        acc3 = wmma_f16(ac, bc3, acc3);                                        \
        ac = an; bc0 = bn0; bc1 = bn1; bc2 = bn2; bc3 = bn3;                   \
    }                                                                          \
    acc0 = wmma_f16(ac, bc0, acc0);                                            \
    acc1 = wmma_f16(ac, bc1, acc1);                                            \
    acc2 = wmma_f16(ac, bc2, acc2);                                            \
    acc3 = wmma_f16(ac, bc3, acc3);

// ---------------- 1) QKV GEMM (16x64 wave tile, pipelined) ------------------
__global__ void __launch_bounds__(256, 1)
qkv_gemm_kernel(const _Float16* __restrict__ xh,
                const _Float16* __restrict__ wh,
                const float* __restrict__ bias,
                _Float16* __restrict__ qh,   // (B,H,T,HD)
                _Float16* __restrict__ kh,   // (B,H,T,HD)
                _Float16* __restrict__ vth)  // (B,H,HD,T)
{
    int lane = threadIdx.x & 31;
    int wave = threadIdx.x >> 5;
    int nbase = blockIdx.x * 64;
    int mtile = blockIdx.y * 128 + wave * 16;

    const _Float16* A0 = xh + (size_t)mtile * D;
    const _Float16* BP = wh + (size_t)nbase * D;

    v8f acc0 = {}, acc1 = {}, acc2 = {}, acc3 = {};
    GEMM_16x64_BODY(A0, BP)

    v8f accs[4] = {acc0, acc1, acc2, acc3};
#pragma unroll
    for (int nt = 0; nt < 4; ++nt) {
        int f = nbase + nt * 16 + (lane & 15);
        int s   = f >> 10;             // 0=q 1=k 2=v
        int rem = f & 1023;
        int h   = rem >> 6;
        int d   = rem & 63;
        float bv = bias[f];
#pragma unroll
        for (int r = 0; r < 8; ++r) {
            int row = mtile + r + 8 * (lane >> 4);
            int b   = row >> 11;
            int t   = row & (T - 1);
            float val = accs[nt][r] + bv;
            size_t bh = (size_t)b * H + h;
            if (s == 0)      qh[(bh * T + t) * HD + d] = (_Float16)val;
            else if (s == 1) kh[(bh * T + t) * HD + d] = (_Float16)val;
            else             vth[(bh * HD + d) * T + t] = (_Float16)val;
        }
    }
}

// ---------------- 2) RoPE (in place), scale folded into q ----------
__global__ void rope_kernel(_Float16* __restrict__ buf, float scale) {
    size_t i = (size_t)blockIdx.x * blockDim.x + threadIdx.x;
    size_t n = (size_t)Bsz * H * T * (HD / 2);
    if (i >= n) return;
    int p  = (int)(i & 31);
    int t  = (int)((i >> 5) & (T - 1));
    size_t bh = i >> 16;               // / (32*T)
    size_t base = ((bh * T) + t) * HD + 2 * p;
    float inv = exp2f(-LOG2_THETA * (float)(2 * p) / (float)HD);
    float ang = (float)t * inv;
    float sn, cs;
    __sincosf(ang, &sn, &cs);
    float e = (float)buf[base], o = (float)buf[base + 1];
    buf[base]     = (_Float16)((e * cs - o * sn) * scale);
    buf[base + 1] = (_Float16)((e * sn + o * cs) * scale);
}

// ---------------- 3) Flash attention: async double-buffered K/V in LDS ------
__global__ void attn_kernel(const _Float16* __restrict__ qh,
                            const _Float16* __restrict__ kh,
                            const _Float16* __restrict__ vth,
                            _Float16* __restrict__ aoh)  // (B,T,D)
{
    __shared__ __align__(32) _Float16 Kt[2][32 * 64];  // K chunk (rows=key, cols=d)
    __shared__ __align__(32) _Float16 Vt[2][64 * 32];  // V^T chunk (rows=d, cols=key)
    __shared__ __align__(32) _Float16 Pt[8][16 * 32];  // per-wave P transpose

    int tid  = threadIdx.x;
    int lane = tid & 31;
    int wave = tid >> 5;
    int qblk = blockIdx.x & 15;             // T/128 = 16
    int bh   = blockIdx.x >> 4;
    int b = bh / H, h = bh % H;

    const _Float16* qb = qh  + (size_t)bh * T * HD;
    const _Float16* kb = kh  + (size_t)bh * T * HD;
    const _Float16* vb = vth + (size_t)bh * HD * T;

    int qrow0 = qblk * 128 + wave * 16;
    v16h qa0 = load_a16x32(qb + (size_t)qrow0 * HD + 0,  HD, lane);
    v16h qa1 = load_a16x32(qb + (size_t)qrow0 * HD + 32, HD, lane);

    // cooperative 16B-per-thread staging of one 32-key chunk (K: 4KB, V: 4KB)
    auto stage = [&](int buf, int kc) {
        copy16B_to_lds(kb + (size_t)kc * HD + tid * 8, &Kt[buf][tid * 8]);
        int d = tid >> 2, c = tid & 3;
        copy16B_to_lds(vb + (size_t)d * T + kc + c * 8, &Vt[buf][d * 32 + c * 8]);
    };
    stage(0, 0);

    float m8[8], l8[8];
#pragma unroll
    for (int r = 0; r < 8; ++r) { m8[r] = -1e30f; l8[r] = 0.0f; }
    v8f o[4] = {};

    for (int kc = 0; kc < T; kc += 32) {
        int cur = (kc >> 5) & 1;
        wait_async_lds();      // own copies into buf[cur] done
        __syncthreads();       // everyone's copies visible; prior reads of buf[1-cur] done
        if (kc + 32 < T) stage(cur ^ 1, kc + 32);   // overlaps with compute below

        const _Float16* kt = Kt[cur];
        v8f s0 = {}, s1 = {};
        s0 = wmma_f16(qa0, load_b32x16(kt, 64, lane), s0);
        s0 = wmma_f16(qa1, load_b32x16(kt + 32, 64, lane), s0);
        s1 = wmma_f16(qa0, load_b32x16(kt + 16 * 64, 64, lane), s1);
        s1 = wmma_f16(qa1, load_b32x16(kt + 16 * 64 + 32, 64, lane), s1);

        // online softmax: row = r + 8*(lane>>4), col = lane&15
        float pr0[8], pr1[8], alpha[8];
#pragma unroll
        for (int r = 0; r < 8; ++r) {
            float mx = fmaxf(s0[r], s1[r]);
            mx = fmaxf(mx, __shfl_xor(mx, 1));
            mx = fmaxf(mx, __shfl_xor(mx, 2));
            mx = fmaxf(mx, __shfl_xor(mx, 4));
            mx = fmaxf(mx, __shfl_xor(mx, 8));
            float mn = fmaxf(m8[r], mx);
            alpha[r] = __expf(m8[r] - mn);
            float p0 = __expf(s0[r] - mn);
            float p1 = __expf(s1[r] - mn);
            float rs = p0 + p1;
            rs += __shfl_xor(rs, 1);
            rs += __shfl_xor(rs, 2);
            rs += __shfl_xor(rs, 4);
            rs += __shfl_xor(rs, 8);
            l8[r] = l8[r] * alpha[r] + rs;
            m8[r] = mn;
            pr0[r] = p0; pr1[r] = p1;
        }
#pragma unroll
        for (int nt = 0; nt < 4; ++nt)
#pragma unroll
            for (int r = 0; r < 8; ++r) o[nt][r] *= alpha[r];

        // P (C-layout) -> LDS -> A-fragment layout (wave-private)
        _Float16* lp = Pt[wave];
#pragma unroll
        for (int r = 0; r < 8; ++r) {
            int row = r + 8 * (lane >> 4);
            lp[row * 32 + (lane & 15)]      = (_Float16)pr0[r];
            lp[row * 32 + 16 + (lane & 15)] = (_Float16)pr1[r];
        }
        v16h pa = load_a16x32(lp, 32, lane);

        const _Float16* vt = Vt[cur];
#pragma unroll
        for (int nt = 0; nt < 4; ++nt)
            o[nt] = wmma_f16(pa, load_b32x16(vt + nt * 16 * 32, 32, lane), o[nt]);
    }

#pragma unroll
    for (int nt = 0; nt < 4; ++nt) {
        int d = nt * 16 + (lane & 15);
#pragma unroll
        for (int r = 0; r < 8; ++r) {
            int t = qrow0 + r + 8 * (lane >> 4);
            float val = o[nt][r] / l8[r];
            aoh[((size_t)b * T + t) * D + h * HD + d] = (_Float16)val;
        }
    }
}

// ---------------- 4) output GEMM (16x64 wave tile, pipelined) ---------------
__global__ void __launch_bounds__(256, 1)
out_gemm_kernel(const _Float16* __restrict__ aoh,
                const _Float16* __restrict__ wh,
                const float* __restrict__ bias,
                float* __restrict__ out)
{
    int lane = threadIdx.x & 31;
    int wave = threadIdx.x >> 5;
    int nbase = blockIdx.x * 64;
    int mtile = blockIdx.y * 128 + wave * 16;

    const _Float16* A0 = aoh + (size_t)mtile * D;
    const _Float16* BP = wh + (size_t)nbase * D;

    v8f acc0 = {}, acc1 = {}, acc2 = {}, acc3 = {};
    GEMM_16x64_BODY(A0, BP)

    v8f accs[4] = {acc0, acc1, acc2, acc3};
#pragma unroll
    for (int nt = 0; nt < 4; ++nt) {
        int n = nbase + nt * 16 + (lane & 15);
        float bv = bias[n];
#pragma unroll
        for (int r = 0; r < 8; ++r) {
            int row = mtile + r + 8 * (lane >> 4);
            out[(size_t)row * D + n] = accs[nt][r] + bv;
        }
    }
}

// ---------------- host launcher ----------------
extern "C" void kernel_launch(void* const* d_in, const int* in_sizes, int n_in,
                              void* d_out, int out_size, void* d_ws, size_t ws_size,
                              hipStream_t stream) {
    const float* x     = (const float*)d_in[0];
    const float* w_qkv = (const float*)d_in[1];
    const float* b_qkv = (const float*)d_in[2];
    const float* w_out = (const float*)d_in[3];
    const float* b_out = (const float*)d_in[4];
    float* out = (float*)d_out;

    _Float16* ws = (_Float16*)d_ws;
    size_t off = 0;
    _Float16* xh    = ws + off; off += (size_t)M_ALL * D;
    _Float16* wqkvh = ws + off; off += (size_t)3 * D * D;
    _Float16* wouth = ws + off; off += (size_t)D * D;
    _Float16* qh    = ws + off; off += (size_t)Bsz * H * T * HD;
    _Float16* kh    = ws + off; off += (size_t)Bsz * H * T * HD;
    _Float16* vth   = ws + off; off += (size_t)Bsz * H * HD * T;
    _Float16* aoh   = ws + off; off += (size_t)M_ALL * D;

    conv_f16_kernel<<<4096, 256, 0, stream>>>(x,     xh,    (size_t)M_ALL * D);
    conv_f16_kernel<<<2048, 256, 0, stream>>>(w_qkv, wqkvh, (size_t)3 * D * D);
    conv_f16_kernel<<<1024, 256, 0, stream>>>(w_out, wouth, (size_t)D * D);

    qkv_gemm_kernel<<<dim3(3 * D / 64, M_ALL / 128), 256, 0, stream>>>(
        xh, wqkvh, b_qkv, qh, kh, vth);

    size_t npair = (size_t)Bsz * H * T * (HD / 2);
    int rblocks = (int)((npair + 255) / 256);
    rope_kernel<<<rblocks, 256, 0, stream>>>(qh, 0.125f);
    rope_kernel<<<rblocks, 256, 0, stream>>>(kh, 1.0f);

    attn_kernel<<<Bsz * H * (T / 128), 256, 0, stream>>>(qh, kh, vth, aoh);

    out_gemm_kernel<<<dim3(D / 64, M_ALL / 128), 256, 0, stream>>>(
        aoh, wouth, b_out, out);
}
